// FinalAttentionQKV_59193239273722
// MI455X (gfx1250) — matrix operand order repaired and checked by reading, not compiled, not run on hardware
//
#include <hip/hip_runtime.h>
#include <hip/hip_bf16.h>

typedef __attribute__((ext_vector_type(16))) __bf16 v16bf;
typedef __attribute__((ext_vector_type(8)))  float  v8f;

#define DIM      256
#define TILE_T   32
#define NCHUNK   8
#define TCHUNK   (4096 / NCHUNK)     // 512 rows per block
#define NTILE    (TCHUNK / TILE_T)   // 16 tiles
#define NTHREADS 256

__device__ __forceinline__ unsigned short f2bf(float f) {
  unsigned u = __float_as_uint(f);
  unsigned r = u + 0x7FFFu + ((u >> 16) & 1u);   // round-to-nearest-even
  return (unsigned short)(r >> 16);
}

// Assemble a v16bf WMMA fragment from two 16-byte LDS reads (ds_load_b128 x2).
__device__ __forceinline__ v16bf ld2q(const unsigned short* p0, const unsigned short* p1) {
  union { v16bf v; uint4 q[2]; } u;
  u.q[0] = *(const uint4*)p0;
  u.q[1] = *(const uint4*)p1;
  return u.v;
}

// One 32x256 (rows x cols) bf16 GEMM tile: out[32][256] = xs[32][256] * W,
// where wt holds W transposed (wt[n*256 + k], bf16). 8 waves, wave w owns
// column tiles {2w, 2w+1}. Accumulate f32 via v_wmma_f32_16x16x32_bf16.
__device__ __forceinline__ void gemm_tile(const unsigned short* __restrict__ xs,
                                          const unsigned short* __restrict__ wt,
                                          float* __restrict__ kvout,
                                          int wave, int lane) {
  const int nlo  = lane & 15;   // n (B/C) or m (A) within tile
  const int half = lane >> 4;
  const int klo  = half * 8;    // A-frag K base within 32
  const int col0 = wave * 2;

  for (int rt = 0; rt < 2; ++rt) {
    v8f c0 = {};
    v8f c1 = {};
    const int m = rt * 16 + nlo;
    for (int ks = 0; ks < 8; ++ks) {
      const int k0 = ks * 32;
      // A 16x32 bf16: elems 0..7 -> k=k0+klo.., elems 8..15 -> k=k0+klo+16..
      const unsigned short* ap = xs + m * DIM + k0 + klo;
      v16bf a = ld2q(ap, ap + 16);
      // B 32x16 bf16: elems 0..7 -> k=k0+klo.., elems 8..15 -> k=k0+klo+16.. for column n.
      const unsigned short* bp0 = wt + (col0 * 16 + nlo) * DIM + k0 + half * 8;
      const unsigned short* bp1 = bp0 + 16 * DIM;
      v16bf b0 = ld2q(bp0, bp0 + 16);
      v16bf b1 = ld2q(bp1, bp1 + 16);
      c0 = __builtin_amdgcn_wmma_f32_16x16x32_bf16(false, a, false, b0, (short)0, c0, false, false);
      c1 = __builtin_amdgcn_wmma_f32_16x16x32_bf16(false, a, false, b1, (short)0, c1, false, false);
    }
    // C layout: reg r -> row = r + 8*half, col = lane&15
    const int rowb = rt * 16 + half * 8;
    for (int r = 0; r < 8; ++r) {
      kvout[(rowb + r) * DIM + (col0    ) * 16 + nlo] = c0[r];
      kvout[(rowb + r) * DIM + (col0 + 1) * 16 + nlo] = c1[r];
    }
  }
}

__global__ void __launch_bounds__(NTHREADS)
attn_chunk_kernel(const float* __restrict__ x,  const float* __restrict__ Wq,
                  const float* __restrict__ bq, const float* __restrict__ Wk,
                  const float* __restrict__ bk, const float* __restrict__ Wv,
                  const float* __restrict__ bv, const float* __restrict__ Wout,
                  float* __restrict__ pacc, float* __restrict__ pm,
                  float* __restrict__ pl) {
  extern __shared__ __align__(16) char smem[];
  unsigned short* wkT  = (unsigned short*)smem;          // 256*256 bf16 (transposed)
  unsigned short* wvT  = wkT + DIM * DIM;                // 256*256 bf16 (transposed)
  unsigned short* xs   = wvT + DIM * DIM;                // 32*256 bf16 tile
  float* kv    = (float*)(xs + TILE_T * DIM);            // 32*256 f32 (k then v)
  float* qv    = kv + TILE_T * DIM;                      // 256  (q + bk folded)
  float* xlast = qv + DIM;                               // 256
  float* bvs   = xlast + DIM;                            // 256
  float* wos   = bvs + DIM;                              // 256
  float* ew    = wos + DIM;                              // 32 (scores -> weights)
  float* scal  = ew + TILE_T;                            // [m, l, corr]

  const int tid  = threadIdx.x;
  const int lane = tid & 31;
  const int wave = tid >> 5;
  const int b    = blockIdx.x / NCHUNK;
  const int ch   = blockIdx.x % NCHUNK;

  // Stage transposed bf16 weights (vectorized f32 reads, scattered b16 stores).
  for (int i4 = tid; i4 < (DIM * DIM) / 4; i4 += NTHREADS) {
    int i = i4 * 4;
    int k = i >> 8, n = i & 255;           // 4 consecutive n, same row k
    float4 a = *(const float4*)&Wk[i];
    float4 c = *(const float4*)&Wv[i];
    wkT[(n + 0) * DIM + k] = f2bf(a.x);
    wkT[(n + 1) * DIM + k] = f2bf(a.y);
    wkT[(n + 2) * DIM + k] = f2bf(a.z);
    wkT[(n + 3) * DIM + k] = f2bf(a.w);
    wvT[(n + 0) * DIM + k] = f2bf(c.x);
    wvT[(n + 1) * DIM + k] = f2bf(c.y);
    wvT[(n + 2) * DIM + k] = f2bf(c.z);
    wvT[(n + 3) * DIM + k] = f2bf(c.w);
  }
  xlast[tid] = x[((size_t)b * 4096 + 4095) * DIM + tid];
  bvs[tid] = bv[tid];
  wos[tid] = Wout[tid];
  if (tid == 0) { scal[0] = -__builtin_inff(); scal[1] = 0.f; scal[2] = 0.f; }
  __syncthreads();

  // qv = x[b, T-1, :] @ Wq + bq + bk   (bk folded in; tiny, Wq stays L2-hot)
  {
    float s = bq[tid] + bk[tid];
    for (int i = 0; i < DIM; ++i) s += xlast[i] * Wq[i * DIM + tid];
    qv[tid] = s;
  }
  __syncthreads();

  float accv = 0.f;  // thread owns output column d = tid
  const size_t xbase = ((size_t)b * 4096 + (size_t)ch * TCHUNK) * DIM;

  for (int tt = 0; tt < NTILE; ++tt) {
    // 1) stage 32x256 x-tile as bf16: float4 global loads, ushort4 LDS stores
    const size_t tb = xbase + (size_t)tt * TILE_T * DIM;
    for (int j = 0; j < (TILE_T * DIM) / (4 * NTHREADS); ++j) {
      int i4 = j * NTHREADS + tid;
      float4 a = *(const float4*)&x[tb + (size_t)i4 * 4];
      ushort4 h;
      h.x = f2bf(a.x); h.y = f2bf(a.y); h.z = f2bf(a.z); h.w = f2bf(a.w);
      *(ushort4*)&xs[i4 * 4] = h;
    }
    // Prefetch next tile into L2/WGP$ while this tile computes
    // (global_prefetch_b8: fire-and-forget, no counter).
    if (tt + 1 < NTILE) {
      const float* np = x + tb + (size_t)TILE_T * DIM;
      __builtin_prefetch(np + tid * 32, 0, 1);   // 256 threads x 128B = 32KB tile
    }
    __syncthreads();

    // 2) k tile = x_tile @ Wk  (WMMA)
    gemm_tile(xs, wkT, kv, wave, lane);
    __syncthreads();

    // 3) e_t = sum_d tanh((q+bk)_d + k_td) * Wout_d  (bout cancels in softmax)
    {
      int row = tid >> 3, seg = tid & 7;
      const float* kr = kv + row * DIM;
      float p = 0.f;
      for (int d = seg * 32; d < seg * 32 + 32; ++d)
        p += tanhf(qv[d] + kr[d]) * wos[d];
      p += __shfl_down(p, 4, 32);
      p += __shfl_down(p, 2, 32);
      p += __shfl_down(p, 1, 32);
      if (seg == 0) ew[row] = p;
    }
    __syncthreads();

    // 4) online softmax state update (wave 0)
    if (tid < 32) {
      float e = ew[tid];
      float tm = e;
      for (int o = 16; o > 0; o >>= 1) tm = fmaxf(tm, __shfl_xor(tm, o, 32));
      float m_old = scal[0];
      float m_new = fmaxf(m_old, tm);
      float w = expf(e - m_new);
      float sw = w;
      for (int o = 16; o > 0; o >>= 1) sw += __shfl_xor(sw, o, 32);
      if (tid == 0) {
        float corr = expf(m_old - m_new);
        scal[1] = scal[1] * corr + sw;
        scal[0] = m_new;
        scal[2] = corr;
      }
      ew[tid] = w;
    }
    __syncthreads();
    accv *= scal[2];

    // 5) v tile = x_tile @ Wv  (WMMA, reuse kv buffer)
    gemm_tile(xs, wvT, kv, wave, lane);
    __syncthreads();

    // 6) acc_d += sum_t w_t * (v_td + bv_d)
    {
      float bvd = bvs[tid];
      for (int t = 0; t < TILE_T; ++t)
        accv += ew[t] * (kv[t * DIM + tid] + bvd);
    }
    __syncthreads();
  }

  pacc[(size_t)blockIdx.x * DIM + tid] = accv;
  if (tid == 0) { pm[blockIdx.x] = scal[0]; pl[blockIdx.x] = scal[1]; }
}

__global__ void __launch_bounds__(NTHREADS)
attn_reduce_kernel(const float* __restrict__ pacc, const float* __restrict__ pm,
                   const float* __restrict__ pl, float* __restrict__ out) {
  __shared__ float sm[NCHUNK], sl[NCHUNK];
  const int tid = threadIdx.x, b = blockIdx.x;
  if (tid < NCHUNK) {
    sm[tid] = pm[b * NCHUNK + tid];
    sl[tid] = pl[b * NCHUNK + tid];
  }
  __syncthreads();
  float M = -__builtin_inff();
  for (int c = 0; c < NCHUNK; ++c) M = fmaxf(M, sm[c]);
  float L = 0.f, o = 0.f;
  for (int c = 0; c < NCHUNK; ++c) {
    float s = expf(sm[c] - M);
    L += sl[c] * s;
    o += s * pacc[(size_t)(b * NCHUNK + c) * DIM + tid];
  }
  out[b * DIM + tid] = o / L;
}

extern "C" void kernel_launch(void* const* d_in, const int* in_sizes, int n_in,
                              void* d_out, int out_size, void* d_ws, size_t ws_size,
                              hipStream_t stream) {
  (void)in_sizes; (void)n_in; (void)out_size; (void)ws_size;
  const float* x    = (const float*)d_in[0];
  const float* Wq   = (const float*)d_in[1];
  const float* bq   = (const float*)d_in[2];
  const float* Wk   = (const float*)d_in[3];
  const float* bk   = (const float*)d_in[4];
  const float* Wv   = (const float*)d_in[5];
  const float* bv   = (const float*)d_in[6];
  const float* Wout = (const float*)d_in[7];
  // d_in[8] (bout): uniform shift of scores -> cancels exactly in softmax.

  const int B = 64, NB = B * NCHUNK;
  float* ws   = (float*)d_ws;
  float* pacc = ws;                         // NB * 256
  float* pm   = pacc + (size_t)NB * DIM;    // NB
  float* pl   = pm + NB;                    // NB

  size_t smem = (size_t)(2 * DIM * DIM + TILE_T * DIM) * sizeof(unsigned short)
              + (size_t)(TILE_T * DIM + 4 * DIM + TILE_T + 8) * sizeof(float);

  attn_chunk_kernel<<<dim3(NB), dim3(NTHREADS), smem, stream>>>(
      x, Wq, bq, Wk, bk, Wv, bv, Wout, pacc, pm, pl);
  attn_reduce_kernel<<<dim3(B), dim3(NTHREADS), 0, stream>>>(
      pacc, pm, pl, (float*)d_out);
}
